// GTS_44049184588234
// MI455X (gfx1250) — matrix operand order repaired and checked by reading, not compiled, not run on hardware
//
#include <hip/hip_runtime.h>
#include <hip/hip_bf16.h>
#include <cstdint>

typedef __attribute__((ext_vector_type(16))) _Float16 v16h;
typedef __attribute__((ext_vector_type(8)))  _Float16 v8h;
typedef __attribute__((ext_vector_type(8)))  float    v8f;
typedef __attribute__((ext_vector_type(4)))  int      v4i;

typedef __attribute__((address_space(1))) v4i gv4i;
typedef __attribute__((address_space(3))) v4i lv4i;

#define N_NODES 2048
#define BATCH   64
#define FCAT    128
#define UNITS   64
#define LDSW    40   // LDS row stride in halves (32 data + 8 pad -> conflict-free b128)

// ---------------- CDNA5 async global->LDS DMA (ASYNCcnt path) ----------------
__device__ __forceinline__ void async_b128(const _Float16* g, _Float16* l) {
  __builtin_amdgcn_global_load_async_to_lds_b128((gv4i*)g, (lv4i*)l, 0, 0);
}
__device__ __forceinline__ void wait_async0() {
  asm volatile("s_wait_asynccnt 0x0" ::: "memory");
}

// ---------------- WMMA fragment loaders (wave32, gfx1250 layouts) ----------------
// A-matrix 16x32 f16: lanes 0-15 = rows, K halves: {hi*8..hi*8+7} and {16+hi*8..}
__device__ __forceinline__ v16h frag_a(const _Float16* base, int row, int lane) {
  const int hi = lane >> 4;
  const _Float16* p = base + row * LDSW + hi * 8;
  v8h lo = *(const v8h*)p;
  v8h hh = *(const v8h*)(p + 16);
  v16h r;
#pragma unroll
  for (int i = 0; i < 8; ++i) { r[i] = lo[i]; r[i + 8] = hh[i]; }
  return r;
}
// B-matrix 32x16 f16 (stored transposed in LDS: row = out column, K contiguous):
// K halves: {hi*8..hi*8+7} and {16+hi*8..}, matching frag_a.
__device__ __forceinline__ v16h frag_b(const _Float16* base, int row, int lane) {
  const int hi = lane >> 4;
  const _Float16* p = base + row * LDSW + hi * 8;
  v8h lo = *(const v8h*)p;
  v8h hh = *(const v8h*)(p + 16);
  v16h r;
#pragma unroll
  for (int i = 0; i < 8; ++i) { r[i] = lo[i]; r[i + 8] = hh[i]; }
  return r;
}

// ---------------- K0: d_inv = 1 / (1 + rowsum(adj)) ----------------
__global__ __launch_bounds__(256) void k_rowsum(const float* __restrict__ adj,
                                                float* __restrict__ dinv) {
  const int m = blockIdx.x, t = threadIdx.x;
  float s = 0.0f;
  for (int i = t; i < N_NODES; i += 256) s += adj[(size_t)m * N_NODES + i];
#pragma unroll
  for (int o = 16; o > 0; o >>= 1) s += __shfl_down(s, o, 32);
  __shared__ float red[8];
  if ((t & 31) == 0) red[t >> 5] = s;
  __syncthreads();
  if (t == 0) {
    float tot = 1.0f;
#pragma unroll
    for (int i = 0; i < 8; ++i) tot += red[i];
    dinv[m] = (tot > 0.0f) ? (1.0f / tot) : 0.0f;
  }
}

// ---------------- K1: adjT_f16[n][m] = (adj[m][n] + (m==n)) * dinv[m] ----------------
__global__ __launch_bounds__(256) void k_adjT(const float* __restrict__ adj,
                                              const float* __restrict__ dinv,
                                              _Float16* __restrict__ adjT) {
  __shared__ float tile[32][33];
  const int tx = threadIdx.x & 31, ty = threadIdx.x >> 5;
  const int mB = blockIdx.x * 32, nB = blockIdx.y * 32;
#pragma unroll
  for (int k = 0; k < 4; ++k) {
    const int m = mB + ty + k * 8, n = nB + tx;
    float v = (adj[(size_t)m * N_NODES + n] + (m == n ? 1.0f : 0.0f)) * dinv[m];
    tile[ty + k * 8][tx] = v;
  }
  __syncthreads();
#pragma unroll
  for (int k = 0; k < 4; ++k) {
    const int n = nB + ty + k * 8, m = mB + tx;
    adjT[(size_t)n * N_NODES + m] = (_Float16)tile[tx][ty + k * 8];
  }
}

// ---------------- K2: pack weights transposed: WT[j][kk], kk = piece*128 + f ----------------
__global__ __launch_bounds__(256) void k_packW(const float* __restrict__ Wru,
                                               const float* __restrict__ Wc,
                                               _Float16* __restrict__ WruT,
                                               _Float16* __restrict__ WcT) {
  const int idx = blockIdx.x * 256 + threadIdx.x;
  if (idx < 128 * 384) {
    const int j = idx / 384, kk = idx % 384, p = kk >> 7, f = kk & 127;
    WruT[idx] = (_Float16)Wru[(f * 3 + p) * 128 + j];
  }
  const int i2 = idx - 128 * 384;
  if (i2 >= 0 && i2 < 64 * 384) {
    const int j = i2 / 384, kk = i2 % 384, p = kk >> 7, f = kk & 127;
    WcT[i2] = (_Float16)Wc[(f * 3 + p) * 64 + j];
  }
}

// ---------------- K3: XH row-major (concat x,h) + XRH x-part ----------------
__global__ __launch_bounds__(256) void k_buildXH(const float* __restrict__ x,
                                                 const float* __restrict__ h,
                                                 _Float16* __restrict__ XH,
                                                 _Float16* __restrict__ XRH) {
  const size_t idx = (size_t)blockIdx.x * 256 + threadIdx.x;
  const int f = (int)(idx & 127);
  const size_t bn = idx >> 7;
  const float v = (f < 64) ? x[bn * 64 + f] : h[bn * 64 + (f - 64)];
  const _Float16 hv = (_Float16)v;
  XH[idx] = hv;
  if (f < 64) XRH[bn * FCAT + f] = hv;
}

// ---------------- K3b: f16 tiled transpose [R=2048][C] -> [C][2048] ----------------
__global__ __launch_bounds__(256) void k_tr16(const _Float16* __restrict__ in,
                                              _Float16* __restrict__ out,
                                              int ldIn, int ldOut,
                                              size_t inBS, size_t outBS) {
  __shared__ alignas(16) _Float16 tl[64][72];
  const int t = threadIdx.x;
  const int r0 = blockIdx.x * 64, c0 = blockIdx.y * 64;
  in += (size_t)blockIdx.z * inBS;
  out += (size_t)blockIdx.z * outBS;
#pragma unroll
  for (int p = 0; p < 2; ++p) {
    const int rl = (t >> 3) + p * 32, c8 = (t & 7) * 8;
    v8h v = *(const v8h*)&in[(size_t)(r0 + rl) * ldIn + c0 + c8];
    *(v8h*)&tl[rl][c8] = v;
  }
  __syncthreads();
#pragma unroll
  for (int p = 0; p < 2; ++p) {
    const int cl = (t >> 3) + p * 32, r8 = (t & 7) * 8;
    v8h v;
#pragma unroll
    for (int e = 0; e < 8; ++e) v[e] = tl[r8 + e][cl];
    *(v8h*)&out[(size_t)(c0 + cl) * ldOut + r0 + r8] = v;
  }
}

// ---------------- K4: diffusion GEMM  out = alpha*(adjT @ X) + beta*Z ----------------
// A = adjT [2048][2048] f16 row-major; B supplied transposed: Bt [B][128][2048] f16.
// Stage-in uses CDNA5 async global->LDS DMA, double buffered; one barrier per chunk.
__global__ __launch_bounds__(256) void diff_gemm(const _Float16* __restrict__ adjT,
                                                 const _Float16* __restrict__ Bt,
                                                 const _Float16* __restrict__ Zrow,
                                                 _Float16* __restrict__ outRow,
                                                 _Float16* __restrict__ outT,
                                                 float alpha, float beta,
                                                 int hasZ, int hasT) {
  __shared__ alignas(16) _Float16 lA[2][128 * LDSW];
  __shared__ alignas(16) _Float16 lB[2][128 * LDSW];
  const int t = threadIdx.x, lane = t & 31, w = t >> 5;
  const int wm = (w >> 1) * 32, wn = (w & 1) * 64;
  const int b = blockIdx.y, nb = blockIdx.x * 128;
  const int lr = t >> 2, lc = (t & 3) * 8;
  const int ldsOff = lr * LDSW + lc;
  const _Float16* aG = adjT + (size_t)(nb + lr) * N_NODES + lc;
  const _Float16* bG = Bt + ((size_t)b * FCAT + lr) * N_NODES + lc;

  v8f acc[2][4];
#pragma unroll
  for (int i = 0; i < 2; ++i)
#pragma unroll
    for (int j = 0; j < 4; ++j)
#pragma unroll
      for (int r = 0; r < 8; ++r) acc[i][j][r] = 0.0f;

  // prologue: async DMA chunk 0 into buffer 0
  async_b128(aG, &lA[0][ldsOff]);
  async_b128(aG + (size_t)64 * N_NODES, &lA[0][ldsOff + 64 * LDSW]);
  async_b128(bG, &lB[0][ldsOff]);
  async_b128(bG + (size_t)64 * N_NODES, &lB[0][ldsOff + 64 * LDSW]);
  wait_async0();
  __syncthreads();

  int buf = 0;
  for (int k = 32; k <= N_NODES; k += 32) {
    const bool last = (k == N_NODES);
    if (!last) {
      // issue DMA for next chunk into the other buffer; overlaps with WMMAs below
      const int nb2 = buf ^ 1;
      async_b128(aG + k, &lA[nb2][ldsOff]);
      async_b128(aG + k + (size_t)64 * N_NODES, &lA[nb2][ldsOff + 64 * LDSW]);
      async_b128(bG + k, &lB[nb2][ldsOff]);
      async_b128(bG + k + (size_t)64 * N_NODES, &lB[nb2][ldsOff + 64 * LDSW]);
    }
    v16h af[2], bf[4];
#pragma unroll
    for (int i = 0; i < 2; ++i) af[i] = frag_a(lA[buf], wm + i * 16 + (lane & 15), lane);
#pragma unroll
    for (int j = 0; j < 4; ++j) bf[j] = frag_b(lB[buf], wn + j * 16 + (lane & 15), lane);
#pragma unroll
    for (int i = 0; i < 2; ++i)
#pragma unroll
      for (int j = 0; j < 4; ++j)
        acc[i][j] = __builtin_amdgcn_wmma_f32_16x16x32_f16(
            false, af[i], false, bf[j], (short)0, acc[i][j], false, false);
    if (!last) {
      wait_async0();
      __syncthreads();
      buf ^= 1;
    }
  }

  const int hi = lane >> 4;
#pragma unroll
  for (int i = 0; i < 2; ++i) {
#pragma unroll
    for (int j = 0; j < 4; ++j) {
      const int fcol = wn + j * 16 + (lane & 15);
      const int nr0 = nb + wm + i * 16 + 8 * hi;
      v8f v = acc[i][j];
      if (hasZ) {
#pragma unroll
        for (int r = 0; r < 8; ++r)
          v[r] = alpha * v[r] +
                 beta * (float)Zrow[((size_t)b * N_NODES + nr0 + r) * FCAT + fcol];
      }
      v8h hv;
#pragma unroll
      for (int r = 0; r < 8; ++r) {
        const _Float16 hval = (_Float16)v[r];
        hv[r] = hval;
        outRow[((size_t)b * N_NODES + nr0 + r) * FCAT + fcol] = hval;
      }
      if (hasT)
        *(v8h*)&outT[((size_t)b * FCAT + fcol) * N_NODES + nr0] = hv;
    }
  }
}

// ---------------- K5: dense ru + sigmoid; emit u (f16) and XRH = [x | r*h] ----------------
__global__ __launch_bounds__(256) void dense_ru(const _Float16* __restrict__ X0,
                                                const _Float16* __restrict__ X1,
                                                const _Float16* __restrict__ X2,
                                                const _Float16* __restrict__ WT,   // [128][384]
                                                const float* __restrict__ bias,    // [128]
                                                const float* __restrict__ hx,
                                                _Float16* __restrict__ XRHrow,     // cols 64..127
                                                _Float16* __restrict__ XRHT,       // rows 64..127
                                                _Float16* __restrict__ U) {
  __shared__ alignas(16) _Float16 lA[2][128 * LDSW];
  __shared__ alignas(16) _Float16 lB[2][128 * LDSW];
  const int t = threadIdx.x, lane = t & 31, w = t >> 5;
  const int wm = (w >> 1) * 32, wn = (w & 1) * 64;
  const size_t rb = (size_t)blockIdx.x * 128;
  const int lr = t >> 2, lc = (t & 3) * 8;
  const int ldsOff = lr * LDSW + lc;
  const _Float16* Xs[3] = {X0, X1, X2};

  v8f acc[2][4];
#pragma unroll
  for (int i = 0; i < 2; ++i)
#pragma unroll
    for (int j = 0; j < 4; ++j)
#pragma unroll
      for (int r = 0; r < 8; ++r) acc[i][j][r] = 0.0f;

  v8h a0 = *(const v8h*)&X0[(rb + lr) * FCAT + lc];
  v8h a1 = *(const v8h*)&X0[(rb + lr + 64) * FCAT + lc];
  v8h b0 = *(const v8h*)&WT[lr * 384 + lc];
  v8h b1 = *(const v8h*)&WT[(lr + 64) * 384 + lc];
  *(v8h*)&lA[0][ldsOff] = a0;
  *(v8h*)&lA[0][ldsOff + 64 * LDSW] = a1;
  *(v8h*)&lB[0][ldsOff] = b0;
  *(v8h*)&lB[0][ldsOff + 64 * LDSW] = b1;
  __syncthreads();

  int buf = 0;
  for (int c = 1; c <= 12; ++c) {
    const bool last = (c == 12);
    if (!last) {
      const _Float16* Xp = Xs[c >> 2];
      const int off = (c & 3) * 32;
      a0 = *(const v8h*)&Xp[(rb + lr) * FCAT + off + lc];
      a1 = *(const v8h*)&Xp[(rb + lr + 64) * FCAT + off + lc];
      b0 = *(const v8h*)&WT[lr * 384 + c * 32 + lc];
      b1 = *(const v8h*)&WT[(lr + 64) * 384 + c * 32 + lc];
    }
    v16h af[2], bf[4];
#pragma unroll
    for (int i = 0; i < 2; ++i) af[i] = frag_a(lA[buf], wm + i * 16 + (lane & 15), lane);
#pragma unroll
    for (int j = 0; j < 4; ++j) bf[j] = frag_b(lB[buf], wn + j * 16 + (lane & 15), lane);
#pragma unroll
    for (int i = 0; i < 2; ++i)
#pragma unroll
      for (int j = 0; j < 4; ++j)
        acc[i][j] = __builtin_amdgcn_wmma_f32_16x16x32_f16(
            false, af[i], false, bf[j], (short)0, acc[i][j], false, false);
    if (!last) {
      __syncthreads();
      const int nb2 = buf ^ 1;
      *(v8h*)&lA[nb2][ldsOff] = a0;
      *(v8h*)&lA[nb2][ldsOff + 64 * LDSW] = a1;
      *(v8h*)&lB[nb2][ldsOff] = b0;
      *(v8h*)&lB[nb2][ldsOff + 64 * LDSW] = b1;
      __syncthreads();
      buf = nb2;
    }
  }

  const int hi = lane >> 4;
#pragma unroll
  for (int i = 0; i < 2; ++i) {
#pragma unroll
    for (int j = 0; j < 4; ++j) {
      const int jc = wn + j * 16 + (lane & 15);
      const size_t r0 = rb + wm + i * 16 + 8 * hi;
      const float bv = bias[jc];
      v8f v = acc[i][j];
      if (jc >= 64) {
        const int ju = jc - 64;
#pragma unroll
        for (int r = 0; r < 8; ++r) {
          const float s = 1.0f / (1.0f + __expf(-(v[r] + bv)));
          U[(r0 + r) * UNITS + ju] = (_Float16)s;
        }
      } else {
        v8h hv;
#pragma unroll
        for (int r = 0; r < 8; ++r) {
          const float s = 1.0f / (1.0f + __expf(-(v[r] + bv)));
          const float rh = s * hx[(r0 + r) * UNITS + jc];
          hv[r] = (_Float16)rh;
          XRHrow[(r0 + r) * FCAT + 64 + jc] = hv[r];
        }
        const size_t bb = r0 >> 11, n0 = r0 & 2047;
        *(v8h*)&XRHT[(bb * FCAT + 64 + jc) * N_NODES + n0] = hv;
      }
    }
  }
}

// ---------------- K7: dense c + tanh + GRU blend -> d_out ----------------
__global__ __launch_bounds__(256) void dense_c(const _Float16* __restrict__ X0,
                                               const _Float16* __restrict__ X1,
                                               const _Float16* __restrict__ X2,
                                               const _Float16* __restrict__ WT,  // [64][384]
                                               const float* __restrict__ bias,   // [64]
                                               const float* __restrict__ hx,
                                               const _Float16* __restrict__ U,
                                               float* __restrict__ out) {
  __shared__ alignas(16) _Float16 lA[2][128 * LDSW];
  __shared__ alignas(16) _Float16 lB[2][64 * LDSW];
  const int t = threadIdx.x, lane = t & 31, w = t >> 5;
  const int wm = w * 16;
  const size_t rb = (size_t)blockIdx.x * 128;
  const int lr = t >> 2, lc = (t & 3) * 8;
  const int ldsOff = lr * LDSW + lc;
  const _Float16* Xs[3] = {X0, X1, X2};

  v8f acc[4];
#pragma unroll
  for (int j = 0; j < 4; ++j)
#pragma unroll
    for (int r = 0; r < 8; ++r) acc[j][r] = 0.0f;

  v8h a0 = *(const v8h*)&X0[(rb + lr) * FCAT + lc];
  v8h a1 = *(const v8h*)&X0[(rb + lr + 64) * FCAT + lc];
  v8h b0 = *(const v8h*)&WT[lr * 384 + lc];
  *(v8h*)&lA[0][ldsOff] = a0;
  *(v8h*)&lA[0][ldsOff + 64 * LDSW] = a1;
  *(v8h*)&lB[0][ldsOff] = b0;
  __syncthreads();

  int buf = 0;
  for (int c = 1; c <= 12; ++c) {
    const bool last = (c == 12);
    if (!last) {
      const _Float16* Xp = Xs[c >> 2];
      const int off = (c & 3) * 32;
      a0 = *(const v8h*)&Xp[(rb + lr) * FCAT + off + lc];
      a1 = *(const v8h*)&Xp[(rb + lr + 64) * FCAT + off + lc];
      b0 = *(const v8h*)&WT[lr * 384 + c * 32 + lc];
    }
    v16h af = frag_a(lA[buf], wm + (lane & 15), lane);
    v16h bf[4];
#pragma unroll
    for (int j = 0; j < 4; ++j) bf[j] = frag_b(lB[buf], j * 16 + (lane & 15), lane);
#pragma unroll
    for (int j = 0; j < 4; ++j)
      acc[j] = __builtin_amdgcn_wmma_f32_16x16x32_f16(
          false, af, false, bf[j], (short)0, acc[j], false, false);
    if (!last) {
      __syncthreads();
      const int nb2 = buf ^ 1;
      *(v8h*)&lA[nb2][ldsOff] = a0;
      *(v8h*)&lA[nb2][ldsOff + 64 * LDSW] = a1;
      *(v8h*)&lB[nb2][ldsOff] = b0;
      __syncthreads();
      buf = nb2;
    }
  }

  const int hi = lane >> 4;
#pragma unroll
  for (int j = 0; j < 4; ++j) {
    const int jc = j * 16 + (lane & 15);
    const size_t r0 = rb + wm + 8 * hi;
    const float bv = bias[jc];
#pragma unroll
    for (int r = 0; r < 8; ++r) {
      const size_t row = r0 + r;
      const float cval = tanhf(acc[j][r] + bv);
      const float uval = (float)U[row * UNITS + jc];
      const float hval = hx[row * UNITS + jc];
      out[row * UNITS + jc] = uval * hval + (1.0f - uval) * cval;
    }
  }
}

extern "C" void kernel_launch(void* const* d_in, const int* in_sizes, int n_in,
                              void* d_out, int out_size, void* d_ws, size_t ws_size,
                              hipStream_t stream) {
  (void)in_sizes; (void)n_in; (void)out_size; (void)ws_size;
  const float* x   = (const float*)d_in[0];
  const float* hx  = (const float*)d_in[1];
  const float* adj = (const float*)d_in[2];
  const float* Wru = (const float*)d_in[3];
  const float* bru = (const float*)d_in[4];
  const float* Wc  = (const float*)d_in[5];
  const float* bc  = (const float*)d_in[6];
  float* out = (float*)d_out;
  char* ws = (char*)d_ws;

  constexpr size_t A = (size_t)BATCH * N_NODES * FCAT * 2;  // 33,554,432 bytes per f16 tensor
  float*    dinv = (float*)(ws + 0);                        // 8 KB
  _Float16* adjT = (_Float16*)(ws + 8192);                  // 8 MB
  _Float16* WruT = (_Float16*)(ws + 8396800);
  _Float16* WcT  = (_Float16*)(ws + 8495104);
  char* base = ws + 8544256;
  _Float16* X0R  = (_Float16*)(base);          // XH row-major
  _Float16* X0T  = (_Float16*)(base + 1 * A);  // XHt, later reused as XRHt
  _Float16* X1R  = (_Float16*)(base + 2 * A);  // later Y1 row
  _Float16* X1T  = (_Float16*)(base + 3 * A);  // later Y1t
  _Float16* X2R  = (_Float16*)(base + 4 * A);  // later Y2 row
  _Float16* XRHR = (_Float16*)(base + 5 * A);  // XRH row-major
  _Float16* U    = (_Float16*)(base + 6 * A);  // u gate f16 [B*N][64]

  const size_t bsRow = (size_t)N_NODES * FCAT;
  const size_t bsT   = (size_t)FCAT * N_NODES;

  k_rowsum<<<N_NODES, 256, 0, stream>>>(adj, dinv);
  k_adjT<<<dim3(N_NODES / 32, N_NODES / 32), 256, 0, stream>>>(adj, dinv, adjT);
  k_packW<<<288, 256, 0, stream>>>(Wru, Wc, WruT, WcT);
  k_buildXH<<<(BATCH * N_NODES * FCAT) / 256, 256, 0, stream>>>(x, hx, X0R, XRHR);
  // XHt
  k_tr16<<<dim3(32, 2, BATCH), 256, 0, stream>>>(X0R, X0T, FCAT, N_NODES, bsRow, bsT);
  // x1 = adjT @ x0          (row + transposed outputs)
  diff_gemm<<<dim3(16, BATCH), 256, 0, stream>>>(adjT, X0T, nullptr, X1R, X1T, 1.0f, 0.0f, 0, 1);
  // XRHt rows 0..63 = x^T   (reuses the X0T slot; X0T dead after previous GEMM)
  k_tr16<<<dim3(32, 1, BATCH), 256, 0, stream>>>(XRHR, X0T, FCAT, N_NODES, bsRow, bsT);
  // x2 = 2*(adjT @ x1) - x0 (row output only)
  diff_gemm<<<dim3(16, BATCH), 256, 0, stream>>>(adjT, X1T, X0R, X2R, nullptr, 2.0f, -1.0f, 1, 0);
  // ru = sigmoid([x0 x1 x2] @ Wru + b); emit u and XRH = [x | r*h] (+XRHt rows 64..127)
  dense_ru<<<1024, 256, 0, stream>>>(X0R, X1R, X2R, WruT, bru, hx, XRHR, X0T, U);
  // y1 = adjT @ y0
  diff_gemm<<<dim3(16, BATCH), 256, 0, stream>>>(adjT, X0T, nullptr, X1R, X1T, 1.0f, 0.0f, 0, 1);
  // y2 = 2*(adjT @ y1) - y0
  diff_gemm<<<dim3(16, BATCH), 256, 0, stream>>>(adjT, X1T, XRHR, X2R, nullptr, 2.0f, -1.0f, 1, 0);
  // c = tanh([y0 y1 y2] @ Wc + b); out = u*h + (1-u)*c
  dense_c<<<1024, 256, 0, stream>>>(XRHR, X1R, X2R, WcT, bc, hx, U, out);
}
